// HCAM_56384330662298
// MI455X (gfx1250) — matrix operand order repaired and checked by reading, not compiled, not run on hardware
//
#include <hip/hip_runtime.h>
#include <hip/hip_bf16.h>
#include <stdint.h>

// Problem constants (fixed by the reference)
#define B_    128
#define N_    256
#define C_    384
#define NH_   6
#define D_    64          // C/NH
#define PD_   24          // C/16
#define TBL_  961         // (2*16-1)^2
#define SCALE_ 0.125f     // D^-0.5

typedef __attribute__((ext_vector_type(16))) __bf16 v16bf;
typedef __attribute__((ext_vector_type(8)))  float  v8f;

union BF16x16 {
  v16bf v;
  unsigned short s[16];
  __bf16 h[16];
  uint4 q[2];
};

union BF16x8 {
  unsigned short s[8];
  __bf16 h[8];
  uint4 q;
};

// Native fptrunc -> v_cvt_pk_bf16_f32 on gfx1250 (RNE, matches reference)
static __device__ __forceinline__ __bf16 f2bf(float f) { return (__bf16)f; }

static __device__ __forceinline__ v8f wmma_bf16(v16bf a, v16bf b, v8f c) {
  // v_wmma_f32_16x16x32_bf16: (neg_a, A, neg_b, B, c_mod, C, reuse_a, reuse_b)
  return __builtin_amdgcn_wmma_f32_16x16x32_bf16(false, a, false, b, (short)0, c,
                                                 false, false);
}

// ---- WMMA operand loaders ---------------------------------------------------
// A (16x32 bf16): lane holds row M=lane&15; two 8-elem K chunks at
//   K = k0 + (lane>>4)*8  and  K = k0 + (lane>>4)*8 + 16.
static __device__ __forceinline__ v16bf load_a_bf16(const unsigned short* row,
                                                    int k0, int hi) {
  BF16x16 r;
  r.q[0] = *(const uint4*)(row + k0 + (hi << 3));
  r.q[1] = *(const uint4*)(row + k0 + (hi << 3) + 16);
  return r.v;
}
// A from fp32 source: load 16 floats, convert in-register (v_cvt_pk_bf16_f32).
static __device__ __forceinline__ v16bf load_a_f32(const float* row, int k0, int hi) {
  const float* p = row + k0 + (hi << 3);
  float4 f0 = *(const float4*)(p + 0);
  float4 f1 = *(const float4*)(p + 4);
  float4 f2 = *(const float4*)(p + 16);
  float4 f3 = *(const float4*)(p + 20);
  BF16x16 r;
  r.h[0]  = f2bf(f0.x); r.h[1]  = f2bf(f0.y); r.h[2]  = f2bf(f0.z); r.h[3]  = f2bf(f0.w);
  r.h[4]  = f2bf(f1.x); r.h[5]  = f2bf(f1.y); r.h[6]  = f2bf(f1.z); r.h[7]  = f2bf(f1.w);
  r.h[8]  = f2bf(f2.x); r.h[9]  = f2bf(f2.y); r.h[10] = f2bf(f2.z); r.h[11] = f2bf(f2.w);
  r.h[12] = f2bf(f3.x); r.h[13] = f2bf(f3.y); r.h[14] = f2bf(f3.z); r.h[15] = f2bf(f3.w);
  return r.v;
}
// B (32x16 bf16): lane holds column N=lane&15 of B = a row of the transposed
// storage; two 8-elem K chunks at K = k0 + (lane>>4)*8 and K = k0 + (lane>>4)*8 + 16.
static __device__ __forceinline__ v16bf load_b_bf16(const unsigned short* row,
                                                    int k0, int hi) {
  BF16x16 r;
  r.q[0] = *(const uint4*)(row + k0 + (hi << 3));
  r.q[1] = *(const uint4*)(row + k0 + (hi << 3) + 16);
  return r.v;
}

// ---- 1) weight transpose + bf16 convert: dst[n][k] = bf16(src[k][n]) --------
__global__ __launch_bounds__(256) void hcam_wt_transpose(
    const float* __restrict__ src, unsigned short* __restrict__ dst, int K, int Nn) {
  int idx = blockIdx.x * 256 + threadIdx.x;
  if (idx >= K * Nn) return;
  int n = idx / K, k = idx % K;
  dst[idx] = __builtin_bit_cast(unsigned short, f2bf(src[k * Nn + n]));
}

// ---- 2) tiny position MLP: 961 rows of (2 -> 24 -> 24 -> 24 -> 6) -----------
static __device__ __forceinline__ void ln_relu24(float* x, const float* g,
                                                 const float* b) {
  float m = 0.f;
#pragma unroll
  for (int j = 0; j < PD_; ++j) m += x[j];
  m *= (1.0f / PD_);
  float v = 0.f;
#pragma unroll
  for (int j = 0; j < PD_; ++j) { float d = x[j] - m; v += d * d; }
  v *= (1.0f / PD_);
  float inv = rsqrtf(v + 1e-5f);
#pragma unroll
  for (int j = 0; j < PD_; ++j) {
    float y = (x[j] - m) * inv * g[j] + b[j];
    x[j] = y > 0.f ? y : 0.f;
  }
}

__global__ __launch_bounds__(256) void hcam_pos_mlp(
    const float* w0, const float* b0,
    const float* g1, const float* be1, const float* w1, const float* bb1,
    const float* g2, const float* be2, const float* w2, const float* bb2,
    const float* g3, const float* be3, const float* w3, const float* bb3,
    float* __restrict__ pos) {
  int i = blockIdx.x * 256 + threadIdx.x;
  if (i >= TBL_) return;
  float gh = (float)(i / 31) - 15.0f;   // 1-H .. H-1
  float gw = (float)(i % 31) - 15.0f;
  float t[PD_], u[PD_];
#pragma unroll
  for (int j = 0; j < PD_; ++j) t[j] = gh * w0[j] + gw * w0[PD_ + j] + b0[j];
  ln_relu24(t, g1, be1);
#pragma unroll
  for (int j = 0; j < PD_; ++j) {
    float a = bb1[j];
#pragma unroll
    for (int kk = 0; kk < PD_; ++kk) a += t[kk] * w1[kk * PD_ + j];
    u[j] = a;
  }
  ln_relu24(u, g2, be2);
#pragma unroll
  for (int j = 0; j < PD_; ++j) {
    float a = bb2[j];
#pragma unroll
    for (int kk = 0; kk < PD_; ++kk) a += u[kk] * w2[kk * PD_ + j];
    t[j] = a;
  }
  ln_relu24(t, g3, be3);
#pragma unroll
  for (int hh = 0; hh < NH_; ++hh) {
    float a = bb3[hh];
#pragma unroll
    for (int kk = 0; kk < PD_; ++kk) a += t[kk] * w3[kk * NH_ + hh];
    pos[i * NH_ + hh] = a;
  }
}

// ---- 3) QKV projection GEMM: (32768 x 384 f32) @ (384 x 768 bf16^T) ---------
// Epilogue splits columns: [0,384) -> out_nd[b][h][n][d] (scaled),
//                          [384,768) -> out_tr[b][h][d][n] (transposed, packed
//                          b128 stores: accumulator rows are contiguous in n).
__global__ __launch_bounds__(256) void hcam_qkv_gemm(
    const float* __restrict__ x, const unsigned short* __restrict__ wT,
    const float* __restrict__ bias, unsigned short* __restrict__ out_nd,
    unsigned short* __restrict__ out_tr, float scale) {
  int wave = (blockIdx.x << 3) + (threadIdx.x >> 5);
  int lane = threadIdx.x & 31;
  int l16 = lane & 15, hi = lane >> 4;
  const int strips = (2 * C_) / 64;          // 12
  int rowTile = wave / strips;
  int colBase = (wave % strips) * 64;
  const float* arow = x + (size_t)(rowTile * 16 + l16) * C_;

  v8f acc[4] = {};
  for (int k0 = 0; k0 < C_; k0 += 32) {
    v16bf a = load_a_f32(arow, k0, hi);
#pragma unroll
    for (int c = 0; c < 4; ++c) {
      v16bf bm = load_b_bf16(wT + (size_t)(colBase + c * 16 + l16) * C_, k0, hi);
      acc[c] = wmma_bf16(a, bm, acc[c]);
    }
  }
  // accumulator element (M = r + 8*hi, Ncol = l16); rows of this lane span
  // 8 consecutive tokens starting at row0.
  int row0 = rowTile * 16 + (hi << 3);
  int bb = row0 >> 8;                 // batch (16-row tile never crosses b)
  int n0 = row0 & (N_ - 1);
#pragma unroll
  for (int c = 0; c < 4; ++c) {
    int col = colBase + c * 16 + l16;
    float bv = bias[col];
    if (col < C_) {                   // q / k half: [b][h][n][d], stride D in n
      int h = col / 64, d = col % 64;
      unsigned short* dst = out_nd + ((size_t)(bb * NH_ + h) * N_ + n0) * D_ + d;
#pragma unroll
      for (int r = 0; r < 8; ++r)
        dst[(size_t)r * D_] =
            __builtin_bit_cast(unsigned short, f2bf((acc[c][r] + bv) * scale));
    } else {                          // v / v_h half: [b][h][d][n], packed b128
      int cc = col - C_;
      int h = cc / 64, d = cc % 64;
      BF16x8 pk;
#pragma unroll
      for (int r = 0; r < 8; ++r) pk.h[r] = f2bf(acc[c][r] + bv);
      *(uint4*)(out_tr + ((size_t)(bb * NH_ + h) * D_ + d) * N_ + n0) = pk.q;
    }
  }
}

// ---- 4) fused attention: S = qk^T, dual bias softmax, P @ V -----------------
__global__ __launch_bounds__(256) void hcam_attn(
    const unsigned short* __restrict__ q, const unsigned short* __restrict__ k,
    const unsigned short* __restrict__ vT, const unsigned short* __restrict__ vhT,
    const float* __restrict__ rpb, const float* __restrict__ pos,
    const int* __restrict__ rpi, unsigned short* __restrict__ ctx1,
    unsigned short* __restrict__ ctx2) {
  __shared__ float S[32][N_];                 // 32 KB fp32 scores
  __shared__ unsigned short P[2][32][N_];     // 32 KB bf16 probs (both softmaxes)

  int bh = blockIdx.x >> 3;
  int n0 = (blockIdx.x & 7) * 32;
  int b = bh / NH_, h = bh % NH_;
  int tid = threadIdx.x;
  int wave = tid >> 5, lane = tid & 31, l16 = lane & 15, hi = lane >> 4;

  const unsigned short* qbase = q + (size_t)bh * N_ * D_;
  const unsigned short* kbase = k + (size_t)bh * N_ * D_;

  // --- stage 0: prefetch this wave's stage-2 V rows (global_prefetch_b8) ----
  // Overlaps the fetch with stage-1 WMMA + softmax; lane pairs (hi=0/1) cover
  // both 256B halves of each 512B row.
#pragma unroll
  for (int tt = 0; tt < 2; ++tt) {
    int t = wave + tt * 8;
    int j = t >> 3;
    int dt = t & 3;
    const unsigned short* vbase = (j == 0 ? vhT : vT) + (size_t)bh * D_ * N_;
    __builtin_prefetch(vbase + (size_t)(dt * 16 + l16) * N_ + (hi << 7), 0, 3);
  }

  // --- stage 1: S tile (32 x 256) via WMMA; wave -> 16 rows x 64 cols -------
  {
    int rt = wave & 1;
    int colq = wave >> 1;
    const unsigned short* arow = qbase + (size_t)(n0 + rt * 16 + l16) * D_;
    v8f acc[4] = {};
    for (int k0 = 0; k0 < D_; k0 += 32) {
      v16bf a = load_a_bf16(arow, k0, hi);
#pragma unroll
      for (int c = 0; c < 4; ++c) {
        v16bf bm =
            load_b_bf16(kbase + (size_t)(colq * 64 + c * 16 + l16) * D_, k0, hi);
        acc[c] = wmma_bf16(a, bm, acc[c]);
      }
    }
#pragma unroll
    for (int c = 0; c < 4; ++c)
#pragma unroll
      for (int r = 0; r < 8; ++r)
        S[rt * 16 + r + (hi << 3)][colq * 64 + c * 16 + l16] = acc[c][r];
  }
  __syncthreads();

  // --- dual softmax: 64 (row, bias) jobs x 4-thread quads -------------------
  {
    int job = tid >> 2;                 // 0..63
    int quad = tid & 3;
    int row = job & 31;
    int j = job >> 5;                   // 0: rpb bias, 1: mlp bias
    const float* btab = j ? pos : rpb;
    const int* rrow = rpi + (size_t)(n0 + row) * N_;
    float vals[64];
    float mx = -1e30f;
#pragma unroll
    for (int i = 0; i < 64; ++i) {
      int m = quad * 64 + i;
      float v = S[row][m] + btab[rrow[m] * NH_ + h];
      vals[i] = v;
      mx = fmaxf(mx, v);
    }
    mx = fmaxf(mx, __shfl_xor(mx, 1, 32));
    mx = fmaxf(mx, __shfl_xor(mx, 2, 32));
    float sum = 0.f;
#pragma unroll
    for (int i = 0; i < 64; ++i) {
      float e = __expf(vals[i] - mx);
      vals[i] = e;
      sum += e;
    }
    sum += __shfl_xor(sum, 1, 32);
    sum += __shfl_xor(sum, 2, 32);
    float inv = 1.0f / sum;
#pragma unroll
    for (int i = 0; i < 64; ++i)
      P[j][row][quad * 64 + i] =
          __builtin_bit_cast(unsigned short, f2bf(vals[i] * inv));
  }
  __syncthreads();

  // --- stage 2: O = P @ V; 16 tiles (2 outputs x 2 row-tiles x 4 d-tiles) ---
#pragma unroll
  for (int tt = 0; tt < 2; ++tt) {
    int t = wave + tt * 8;
    int j = t >> 3;
    int rt = (t & 7) >> 2;
    int dt = t & 3;
    // j==0: a1 (rpb softmax) pairs with v_h; j==1: a2 (mlp softmax) with v.
    const unsigned short* vbase = (j == 0 ? vhT : vT) + (size_t)bh * D_ * N_;
    const unsigned short* prow = &P[j][rt * 16 + l16][0];
    v8f acc = {};
    for (int k0 = 0; k0 < N_; k0 += 32) {
      v16bf a = load_a_bf16(prow, k0, hi);            // ds_load_b128
      v16bf bm = load_b_bf16(vbase + (size_t)(dt * 16 + l16) * N_, k0, hi);
      acc = wmma_bf16(a, bm, acc);
    }
    unsigned short* ctx = (j == 0) ? ctx1 : ctx2;
    int cd = h * 64 + dt * 16 + l16;
#pragma unroll
    for (int r = 0; r < 8; ++r) {
      int n = n0 + rt * 16 + r + (hi << 3);
      ctx[(size_t)(b * N_ + n) * C_ + cd] =
          __builtin_bit_cast(unsigned short, f2bf(acc[r]));
    }
  }
}

// ---- 5) output projection GEMM: ctx(bf16) @ projT(bf16) + bias -> f32 ------
__global__ __launch_bounds__(256) void hcam_out_gemm(
    const unsigned short* __restrict__ ctx, const unsigned short* __restrict__ wT,
    const float* __restrict__ bias, float* __restrict__ out) {
  int wave = (blockIdx.x << 3) + (threadIdx.x >> 5);
  int lane = threadIdx.x & 31;
  int l16 = lane & 15, hi = lane >> 4;
  const int strips = C_ / 64;                // 6
  int rowTile = wave / strips;
  int colBase = (wave % strips) * 64;
  const unsigned short* arow = ctx + (size_t)(rowTile * 16 + l16) * C_;

  v8f acc[4] = {};
  for (int k0 = 0; k0 < C_; k0 += 32) {
    v16bf a = load_a_bf16(arow, k0, hi);
#pragma unroll
    for (int c = 0; c < 4; ++c) {
      v16bf bm = load_b_bf16(wT + (size_t)(colBase + c * 16 + l16) * C_, k0, hi);
      acc[c] = wmma_bf16(a, bm, acc[c]);
    }
  }
#pragma unroll
  for (int c = 0; c < 4; ++c) {
    int col = colBase + c * 16 + l16;
    float bv = bias[col];
#pragma unroll
    for (int r = 0; r < 8; ++r) {
      int row = rowTile * 16 + r + (hi << 3);
      out[(size_t)row * C_ + col] = acc[c][r] + bv;
    }
  }
}

// ---- host-side launch -------------------------------------------------------
extern "C" void kernel_launch(void* const* d_in, const int* in_sizes, int n_in,
                              void* d_out, int out_size, void* d_ws, size_t ws_size,
                              hipStream_t stream) {
  const float* x1      = (const float*)d_in[0];
  const float* x2      = (const float*)d_in[1];
  const float* qv_w    = (const float*)d_in[2];
  const float* qv_b    = (const float*)d_in[3];
  const float* kv_w    = (const float*)d_in[4];
  const float* kv_b    = (const float*)d_in[5];
  const float* p1_w    = (const float*)d_in[6];
  const float* p1_b    = (const float*)d_in[7];
  const float* p2_w    = (const float*)d_in[8];
  const float* p2_b    = (const float*)d_in[9];
  const float* rpb     = (const float*)d_in[10];
  const float* pp_w    = (const float*)d_in[11];
  const float* pp_b    = (const float*)d_in[12];
  const float* ln1_g   = (const float*)d_in[13];
  const float* ln1_b   = (const float*)d_in[14];
  const float* l1_w    = (const float*)d_in[15];
  const float* l1_b    = (const float*)d_in[16];
  const float* ln2_g   = (const float*)d_in[17];
  const float* ln2_b   = (const float*)d_in[18];
  const float* l2_w    = (const float*)d_in[19];
  const float* l2_b    = (const float*)d_in[20];
  const float* ln3_g   = (const float*)d_in[21];
  const float* ln3_b   = (const float*)d_in[22];
  const float* l3_w    = (const float*)d_in[23];
  const float* l3_b    = (const float*)d_in[24];
  const int*   rpi     = (const int*)d_in[25];

  const size_t BNC  = (size_t)B_ * N_ * C_;        // 12,582,912
  const size_t BHND = (size_t)B_ * NH_ * N_ * D_;  // == BNC

  // carve workspace (all scratch is written before read within this call)
  size_t off = 0;
  auto carve = [&](size_t bytes) -> void* {
    void* p = (char*)d_ws + off;
    off += (bytes + 255) & ~(size_t)255;
    return p;
  };
  unsigned short* wqvT = (unsigned short*)carve((size_t)2 * C_ * C_ * 2);
  unsigned short* wkvT = (unsigned short*)carve((size_t)2 * C_ * C_ * 2);
  unsigned short* p1T  = (unsigned short*)carve((size_t)C_ * C_ * 2);
  unsigned short* p2T  = (unsigned short*)carve((size_t)C_ * C_ * 2);
  float*          pos  = (float*)carve((size_t)TBL_ * NH_ * 4);
  unsigned short* q_s  = (unsigned short*)carve(BHND * 2);
  unsigned short* k_s  = (unsigned short*)carve(BHND * 2);
  unsigned short* vT   = (unsigned short*)carve(BHND * 2);
  unsigned short* vhT  = (unsigned short*)carve(BHND * 2);
  unsigned short* ctx1 = (unsigned short*)carve(BNC * 2);
  unsigned short* ctx2 = (unsigned short*)carve(BNC * 2);
  (void)ws_size; (void)in_sizes; (void)n_in; (void)out_size;

  // 1) weight prep
  hcam_wt_transpose<<<(2 * C_ * C_ + 255) / 256, 256, 0, stream>>>(qv_w, wqvT, C_, 2 * C_);
  hcam_wt_transpose<<<(2 * C_ * C_ + 255) / 256, 256, 0, stream>>>(kv_w, wkvT, C_, 2 * C_);
  hcam_wt_transpose<<<(C_ * C_ + 255) / 256, 256, 0, stream>>>(p1_w, p1T, C_, C_);
  hcam_wt_transpose<<<(C_ * C_ + 255) / 256, 256, 0, stream>>>(p2_w, p2T, C_, C_);

  // 2) position-bias MLP (961 rows)
  hcam_pos_mlp<<<(TBL_ + 255) / 256, 256, 0, stream>>>(
      pp_w, pp_b, ln1_g, ln1_b, l1_w, l1_b, ln2_g, ln2_b, l2_w, l2_b,
      ln3_g, ln3_b, l3_w, l3_b, pos);

  // 3) projections: 32768x768 output -> 2048*12 wave-jobs -> 3072 blocks
  hcam_qkv_gemm<<<3072, 256, 0, stream>>>(x1, wqvT, qv_b, q_s, vT, SCALE_);
  hcam_qkv_gemm<<<3072, 256, 0, stream>>>(x2, wkvT, kv_b, k_s, vhT, 1.0f);

  // 4) fused attention: B*NH*(N/32) = 6144 blocks
  hcam_attn<<<B_ * NH_ * (N_ / 32), 256, 0, stream>>>(q_s, k_s, vT, vhT, rpb, pos,
                                                      rpi, ctx1, ctx2);

  // 5) output projections: 2048*6 wave-jobs -> 1536 blocks each
  float* o1 = (float*)d_out;
  float* o2 = (float*)d_out + BNC;
  hcam_out_gemm<<<1536, 256, 0, stream>>>(ctx1, p1T, p1_b, o1);
  hcam_out_gemm<<<1536, 256, 0, stream>>>(ctx2, p2T, p2_b, o2);
}
